// LRSelfAttentionBlock_51926154609243
// MI455X (gfx1250) — matrix operand order repaired and checked by reading, not compiled, not run on hardware
//
#include <hip/hip_runtime.h>
#include <hip/hip_bf16.h>
#include <cstdint>
#include <cstddef>

#define KB 4
#define KN 4096
#define KC 8

typedef __attribute__((ext_vector_type(16))) _Float16 v16h;
typedef __attribute__((ext_vector_type(8)))  float    v8f;
typedef int v4i_ __attribute__((vector_size(16)));

union V16HU { v16h v; uint4 q[2]; };

// ---------------- workspace layout (byte offsets, all 16B aligned) -------------
#define OFF_W3J   0u                       // 14 * 2197 floats  (123,032 B)
#define OFF_PB    123136u                  // KN floats         (16,384 B)
#define OFF_Q     139520u                  // KB*KN*32 halves   (1,048,576 B)
#define OFF_VALST 1188096u                 // KB*176*KN halves  (5,767,168 B)
#define OFF_H     6955264u                 // KB*KN*176 floats  (11,534,336 B)
#define OFF_CTX   18489600u                // KB*KN*176 floats  (11,534,336 B)

// paths: 6 VAL then 8 OUT
__constant__ int c_pl1[14] = {4,4,6,4,6,6, 4,4,6,6,4,4,6,6};
__constant__ int c_pl2[14] = {0,2,2,2,0,2, 4,6,4,6,4,6,4,6};
__constant__ int c_pl3[14] = {4,4,4,6,6,6, 4,4,4,4,6,6,6,6};

// ---------------- CDNA5 feature probes ----------------------------------------
#if __has_builtin(__builtin_amdgcn_global_load_async_to_lds_b128)
#define USE_ASYNC_LDS 1
__device__ inline void async_cp16(const _Float16* src, _Float16* dst){
  __builtin_amdgcn_global_load_async_to_lds_b128(
      (v4i_*)const_cast<_Float16*>(src), (v4i_*)dst, 0, 0);
}
#endif

__device__ inline void wait_async0(){
#if __has_builtin(__builtin_amdgcn_s_wait_asynccnt)
  __builtin_amdgcn_s_wait_asynccnt(0);
#else
  asm volatile("s_wait_asynccnt 0" ::: "memory");
#endif
}

// xor-k permutation within 16-lane rows via v_permlane16_b32 (VALU, no LDS)
#if __has_builtin(__builtin_amdgcn_permlane16)
#define USE_PERMLANE 1
__device__ inline float pl16(float v, unsigned lo, unsigned hi){
  int iv = __float_as_int(v);
  int r = __builtin_amdgcn_permlane16(iv, iv, lo, hi, false, false);
  return __int_as_float(r);
}
__device__ inline float red16_max(float x){
  x = fmaxf(x, pl16(x, 0x67452301u, 0xEFCDAB89u));  // xor 1
  x = fmaxf(x, pl16(x, 0x54761032u, 0xDCFE98BAu));  // xor 2
  x = fmaxf(x, pl16(x, 0x32107654u, 0xBA98FEDCu));  // xor 4
  x = fmaxf(x, pl16(x, 0xFEDCBA98u, 0x76543210u));  // xor 8
  return x;
}
__device__ inline float red16_sum(float x){
  x += pl16(x, 0x67452301u, 0xEFCDAB89u);
  x += pl16(x, 0x54761032u, 0xDCFE98BAu);
  x += pl16(x, 0x32107654u, 0xBA98FEDCu);
  x += pl16(x, 0xFEDCBA98u, 0x76543210u);
  return x;
}
#else
__device__ inline float red16_max(float x){
  for (int mk=1; mk<16; mk<<=1) x = fmaxf(x, __shfl_xor(x, mk, 32));
  return x;
}
__device__ inline float red16_sum(float x){
  for (int mk=1; mk<16; mk<<=1) x += __shfl_xor(x, mk, 32);
  return x;
}
#endif

// ====================== Wigner-3j init (computed on device) ====================
__device__ inline double dfact(int n){ double r=1.0; for(int i=2;i<=n;++i) r*=(double)i; return r; }

__device__ double su2_cg(int l1,int l2,int l3,int m1,int m2,int m3){
  if (m1+m2 != m3) return 0.0;
  double pref = sqrt((2.0*l3+1.0)*dfact(l3+l1-l2)*dfact(l3-l1+l2)*dfact(l1+l2-l3)/dfact(l1+l2+l3+1));
  pref *= sqrt(dfact(l3+m3)*dfact(l3-m3)*dfact(l1-m1)*dfact(l1+m1)*dfact(l2-m2)*dfact(l2+m2));
  double s = 0.0;
  for (int k=0;k<=l1+l2-l3;++k){
    int d0=k, d1=l1+l2-l3-k, d2=l1-m1-k, d3=l2+m2-k, d4=l3-l2+m1+k, d5=l3-l1-m2+k;
    if (d0<0||d1<0||d2<0||d3<0||d4<0||d5<0) continue;
    double denom = dfact(d0)*dfact(d1)*dfact(d2)*dfact(d3)*dfact(d4)*dfact(d5);
    s += ((k&1) ? -1.0 : 1.0)/denom;
  }
  return pref*s;
}

__device__ void fillQ(int l, double* qr, double* qi){
  int n = 2*l+1;
  for (int a=0;a<n*n;++a){ qr[a]=0.0; qi[a]=0.0; }
  const double rs2 = 0.70710678118654752440;
  for (int m=-l;m<0;++m){
    qr[(l+m)*n + (l-m)] = rs2;      // q[l+m, l+|m|] = 1/sqrt2
    qi[(l+m)*n + (l+m)] = -rs2;     // q[l+m, l-|m|] = -i/sqrt2
  }
  qr[l*n + l] = 1.0;
  for (int m=1;m<=l;++m){
    double sgn = (m&1) ? -1.0 : 1.0;
    qr[(l+m)*n + (l+m)] = sgn*rs2;
    qi[(l+m)*n + (l-m)] = sgn*rs2;
  }
  if ((l/2)&1){ for (int a=0;a<n*n;++a){ qr[a]=-qr[a]; qi[a]=-qi[a]; } }
}

__global__ void __launch_bounds__(256) w3j_init_kernel(float* __restrict__ w3j){
  __shared__ double sCc[13*13*13];
  __shared__ double sQ1r[169], sQ1i[169], sQ2r[169], sQ2i[169], sQ3r[169], sQ3i[169];
  __shared__ double sRed[256];
  __shared__ double sScale;
  const int p = blockIdx.x, tid = threadIdx.x;
  const int l1 = c_pl1[p], l2 = c_pl2[p], l3 = c_pl3[p];
  const int n1 = 2*l1+1, n2 = 2*l2+1, n3 = 2*l3+1;
  if (tid == 0) fillQ(l1, sQ1r, sQ1i);
  else if (tid == 1) fillQ(l2, sQ2r, sQ2i);
  else if (tid == 2) fillQ(l3, sQ3r, sQ3i);
  const int tot = n1*n2*n3;
  for (int e=tid; e<tot; e+=256){
    int i = e/(n2*n3), rem = e%(n2*n3), k = rem/n3, m = rem%n3;
    sCc[e] = su2_cg(l1,l2,l3, i-l1, k-l2, m-l3);
  }
  __syncthreads();
  float* out = w3j + p*2197;
  double ss = 0.0;
  for (int e=tid; e<tot; e+=256){
    int a = e/(n2*n3), rem = e%(n2*n3), bb = rem/n3, c = rem%n3;
    double accr = 0.0;
    for (int i=0;i<n1;++i){
      double q1r = sQ1r[i*n1+a], q1i = sQ1i[i*n1+a];
      if (q1r==0.0 && q1i==0.0) continue;
      for (int k=0;k<n2;++k){
        double q2r = sQ2r[k*n2+bb], q2i = sQ2i[k*n2+bb];
        if (q2r==0.0 && q2i==0.0) continue;
        double tr = q1r*q2r - q1i*q2i, ti = q1r*q2i + q1i*q2r;
        for (int m=0;m<n3;++m){
          double cc = sCc[(i*n2+k)*n3 + m];
          if (cc==0.0) continue;
          double q3r = sQ3r[m*n3+c], q3i = -sQ3i[m*n3+c]; // conj
          accr += cc*(tr*q3r - ti*q3i);
        }
      }
    }
    out[e] = (float)accr;
    ss += accr*accr;
  }
  sRed[tid] = ss;
  __syncthreads();
  for (int s=128;s>0;s>>=1){ if (tid<s) sRed[tid]+=sRed[tid+s]; __syncthreads(); }
  if (tid == 0){
    double nrm = sqrt(sRed[0]);
    double pscale = (p < 6) ? sqrt((2.0*l3+1.0)/(3.0*KC)) : sqrt((2.0*l3+1.0)/(4.0*KC*KC));
    sScale = pscale/nrm;
  }
  __syncthreads();
  float sc = (float)sScale;
  for (int e=tid; e<tot; e+=256) out[e] *= sc;
}

// ====================== prep: q / pb / h / valsT (per node) ====================
__global__ void __launch_bounds__(64) prep_kernel(
    const float* __restrict__ feat, const float* __restrict__ sh,
    const float* __restrict__ log_s, const float* __restrict__ pos_w,
    const float* __restrict__ pos_b, const float* __restrict__ w_lin_in,
    const float* __restrict__ w_val, const float* __restrict__ w3j,
    _Float16* __restrict__ qout, float* __restrict__ pb,
    _Float16* __restrict__ valsT, float* __restrict__ hbuf)
{
  __shared__ float sF[22], sY[6], sScal[4];
  __shared__ float sH[176], sT[169], sG[104], sV4[72], sV6[104];
  const int tid = threadIdx.x;
  const int node = blockIdx.x, b = node/KN, n = node%KN;
  if (tid < 22) sF[tid] = feat[(size_t)node*22 + tid];
  if (tid < 6)  sY[tid] = sh[n*6 + tid];
  __syncthreads();
  if (tid == 0){
    float a4=0.f, a6=0.f;
    for (int i=0;i<9;++i)  a4 += sF[i]*sF[i];
    for (int i=9;i<22;++i) a6 += sF[i]*sF[i];
    sScal[0] = fmaxf(sqrtf(a4), 1e-12f);
    sScal[1] = fmaxf(sqrtf(a6), 1e-12f);
    sScal[2] = __expf(0.5f*log_s[0]);   // sqrt(s4)
    sScal[3] = __expf(0.5f*log_s[1]);   // sqrt(s6)
    if (b == 0){
      float acc = pos_b[0];
      for (int i=0;i<6;++i) acc += sY[i]*pos_w[i];
      pb[n] = acc;
    }
  }
  __syncthreads();
  if (tid < 32){                        // packed f16 q row (22 used, pad to 32)
    float v = 0.f;
    if (tid < 9)       v = sScal[2]*sF[tid]/sScal[0];
    else if (tid < 22) v = sScal[3]*sF[tid]/sScal[1];
    qout[(size_t)node*32 + tid] = (_Float16)v;
  }
  for (int e=tid; e<176; e+=64){        // h: e = u*22 + i (i<9 -> l=4, else l=6)
    int u = e/22, i = e%22;
    float w = (i < 9) ? w_lin_in[u] : w_lin_in[8+u];
    float h = sF[i]*w;
    sH[e] = h;
    hbuf[(size_t)node*176 + e] = h;
  }
  for (int e=tid; e<72;  e+=64) sV4[e]=0.f;
  for (int e=tid; e<104; e+=64) sV6[e]=0.f;
  __syncthreads();
  for (int p=0; p<6; ++p){
    const int l1=c_pl1[p], l2=c_pl2[p], l3=c_pl3[p];
    const int ni=2*l1+1, nj=2*l2+1, nk=2*l3+1;
    const float* W = w3j + p*2197;
    for (int e=tid; e<ni*nk; e+=64){    // T[i,k] = sum_j y[j] W[i,j,k]
      int i=e/nk, k=e%nk;
      float t = 0.f;
      for (int j=0;j<nj;++j){
        float y = (l2==0) ? sY[0] : sY[1+j];
        t += y * W[(i*nj+j)*nk + k];
      }
      sT[e] = t;
    }
    __syncthreads();
    for (int e=tid; e<KC*nk; e+=64){    // G[u,k] = sum_i h[u,i] T[i,k]
      int u=e/nk, k=e%nk;
      int hoff = u*22 + ((l1==4)?0:9);
      float g = 0.f;
      for (int i=0;i<ni;++i) g += sH[hoff+i]*sT[i*nk+k];
      sG[e] = g;
    }
    __syncthreads();
    for (int e=tid; e<KC*nk; e+=64){    // v[w,k] += sum_u wval[u,w] G[u,k]
      int w=e/nk, k=e%nk;
      float acc = 0.f;
      for (int u=0;u<KC;++u) acc += w_val[p*64 + u*8 + w]*sG[u*nk+k];
      if (l3==4) sV4[w*9+k] += acc; else sV6[w*13+k] += acc;
    }
    __syncthreads();
  }
  for (int d=tid; d<176; d+=64){        // transposed vals: (b, d, n) for B-operand staging
    float v = (d < 72) ? sV4[d] : sV6[d-72];
    valsT[((size_t)b*176 + d)*KN + n] = (_Float16)v;
  }
}

// ====================== flash attention (WMMA core) ============================
__global__ void __launch_bounds__(256, 1) flash_kernel(
    const _Float16* __restrict__ qg, const float* __restrict__ pb,
    const _Float16* __restrict__ valsT, float* __restrict__ ctx)
{
  __shared__ __align__(16) _Float16 sQcol[2][32*32];    // 4 KB
  __shared__ __align__(16) _Float16 sVal[2][176*32];    // 22 KB
  __shared__ __align__(16) _Float16 sP[8][16*32];       // 8 KB

  const int tid  = threadIdx.x;
  const int lane = tid & 31;
  const int wv   = tid >> 5;
  const int r    = lane & 15;
  const int hh   = lane >> 4;
  const int b    = blockIdx.y;
  const int rowBase = blockIdx.x*128 + wv*16;
  const _Float16* qb = qg    + (size_t)b*KN*32;
  const _Float16* vb = valsT + (size_t)b*176*KN;

  // A operand: 16 q rows for this wave (K contiguous per half: 0-15 vs 16-31)
  v16h aQ;
  {
    V16HU u;
    const _Float16* rp = qb + (size_t)(rowBase + r)*32;
    u.q[0] = *(const uint4*)(rp + hh*16);
    u.q[1] = *(const uint4*)(rp + hh*16 + 8);
    aQ = u.v;
  }
  float prow[8];
  #pragma unroll
  for (int i=0;i<8;++i) prow[i] = pb[rowBase + i + 8*hh];

  float m_run[8], l_run[8];
  v8f acc[11];
  #pragma unroll
  for (int i=0;i<8;++i){ m_run[i] = -3.0e38f; l_run[i] = 0.f; }
  #pragma unroll
  for (int t=0;t<11;++t)
    #pragma unroll
    for (int i=0;i<8;++i) acc[t][i] = 0.f;

  // ---- stage tile 0 (832 x 16B chunks: 128 for q-cols, 704 for valsT) ----
#ifdef USE_ASYNC_LDS
  #pragma unroll
  for (int k=0;k<4;++k){
    int idx = tid + k*256;
    if (idx < 832){
      if (idx < 128){
        int row = idx>>2, part = idx&3;
        async_cp16(qb + (size_t)row*32 + part*8, &sQcol[0][row*32 + part*8]);
      } else {
        int j = idx-128, d = j>>2, part = j&3;
        async_cp16(vb + (size_t)d*KN + part*8, &sVal[0][d*32 + part*8]);
      }
    }
  }
  wait_async0();
#else
  #pragma unroll
  for (int k=0;k<4;++k){
    int idx = tid + k*256;
    if (idx < 832){
      if (idx < 128){
        int row = idx>>2, part = idx&3;
        *(uint4*)(&sQcol[0][row*32 + part*8]) = *(const uint4*)(qb + (size_t)row*32 + part*8);
      } else {
        int j = idx-128, d = j>>2, part = j&3;
        *(uint4*)(&sVal[0][d*32 + part*8]) = *(const uint4*)(vb + (size_t)d*KN + part*8);
      }
    }
  }
#endif
  __syncthreads();

  for (int it=0; it<128; ++it){
    const int buf = it & 1;
    const int colBase = it*32;
    const int nColBase = colBase + 32;

#ifdef USE_ASYNC_LDS
    // kick off async staging of the next tile straight into LDS (other buffer)
    if (it+1 < 128){
      #pragma unroll
      for (int k=0;k<4;++k){
        int idx = tid + k*256;
        if (idx < 832){
          if (idx < 128){
            int row = idx>>2, part = idx&3;
            async_cp16(qb + (size_t)(nColBase+row)*32 + part*8, &sQcol[buf^1][row*32 + part*8]);
          } else {
            int j = idx-128, d = j>>2, part = j&3;
            async_cp16(vb + (size_t)d*KN + nColBase + part*8, &sVal[buf^1][d*32 + part*8]);
          }
        }
      }
      if (it+2 < 128)
        __builtin_prefetch((const void*)(vb + (size_t)(tid%176)*KN + colBase + 64), 0, 3);
    }
#else
    uint4 tmp[4];
    if (it+1 < 128){
      #pragma unroll
      for (int k=0;k<4;++k){
        int idx = tid + k*256;
        if (idx < 832){
          if (idx < 128){
            int row = idx>>2, part = idx&3;
            tmp[k] = *(const uint4*)(qb + (size_t)(nColBase+row)*32 + part*8);
          } else {
            int j = idx-128, d = j>>2, part = j&3;
            tmp[k] = *(const uint4*)(vb + (size_t)d*KN + nColBase + part*8);
          }
        }
      }
      if (it+2 < 128)
        __builtin_prefetch((const void*)(vb + (size_t)(tid%176)*KN + colBase + 64), 0, 3);
    }
#endif

    // two 16x16 score tiles via WMMA (B swizzle: lane=column, K contiguous per half)
    v8f sc0, sc1;
    {
      v8f z;
      #pragma unroll
      for (int i=0;i<8;++i) z[i] = 0.f;
      V16HU u0, u1;
      const _Float16* c0 = &sQcol[buf][(r)*32];
      u0.q[0] = *(const uint4*)(c0 + hh*16);
      u0.q[1] = *(const uint4*)(c0 + hh*16 + 8);
      sc0 = __builtin_amdgcn_wmma_f32_16x16x32_f16(false, aQ, false, u0.v, (short)0, z, false, false);
      const _Float16* c1 = &sQcol[buf][(16 + r)*32];
      u1.q[0] = *(const uint4*)(c1 + hh*16);
      u1.q[1] = *(const uint4*)(c1 + hh*16 + 8);
      sc1 = __builtin_amdgcn_wmma_f32_16x16x32_f16(false, aQ, false, u1.v, (short)0, z, false, false);
    }
    float pc0 = pb[colBase + r];
    float pc1 = pb[colBase + 16 + r];
    #pragma unroll
    for (int i=0;i<8;++i){ sc0[i] += prow[i] + pc0; sc1[i] += prow[i] + pc1; }

    // online softmax (rows live across 16-lane halves -> permlane16 reductions)
    float scl[8], rs[8];
    #pragma unroll
    for (int i=0;i<8;++i){
      float mt = red16_max(fmaxf(sc0[i], sc1[i]));
      float mn = fmaxf(m_run[i], mt);
      scl[i] = __expf(m_run[i]-mn);
      m_run[i] = mn;
      sc0[i] = __expf(sc0[i]-mn);
      sc1[i] = __expf(sc1[i]-mn);
      rs[i] = sc0[i]+sc1[i];
    }
    #pragma unroll
    for (int i=0;i<8;++i) rs[i] = red16_sum(rs[i]);
    #pragma unroll
    for (int i=0;i<8;++i) l_run[i] = l_run[i]*scl[i] + rs[i];
    #pragma unroll
    for (int t=0;t<11;++t)
      #pragma unroll
      for (int i=0;i<8;++i) acc[t][i] *= scl[i];

    // P tile (16x32 f16) through wave-private LDS to re-swizzle C/D -> A layout
    #pragma unroll
    for (int i=0;i<8;++i){
      sP[wv][(i+8*hh)*32 + r]      = (_Float16)sc0[i];
      sP[wv][(i+8*hh)*32 + 16 + r] = (_Float16)sc1[i];
    }
    asm volatile("s_wait_dscnt 0" ::: "memory");   // in-wave LDS store->load fence
    v16h aP;
    {
      V16HU u;
      const _Float16* rp = &sP[wv][r*32];
      u.q[0] = *(const uint4*)(rp + hh*16);
      u.q[1] = *(const uint4*)(rp + hh*16 + 8);
      aP = u.v;
    }
    // acc += P @ V over 11 sixteen-wide chunks of the 176-dim values
    #pragma unroll
    for (int t=0;t<11;++t){
      V16HU u;
      const _Float16* vp = &sVal[buf][(t*16 + r)*32];
      u.q[0] = *(const uint4*)(vp + hh*16);
      u.q[1] = *(const uint4*)(vp + hh*16 + 8);
      acc[t] = __builtin_amdgcn_wmma_f32_16x16x32_f16(false, aP, false, u.v, (short)0, acc[t], false, false);
    }

#ifdef USE_ASYNC_LDS
    if (it+1 < 128) wait_async0();   // our async writes to the other buffer done
    __syncthreads();                 // visible to all waves before next iteration
#else
    __syncthreads();
    if (it+1 < 128){
      #pragma unroll
      for (int k=0;k<4;++k){
        int idx = tid + k*256;
        if (idx < 832){
          if (idx < 128){
            int row = idx>>2, part = idx&3;
            *(uint4*)(&sQcol[buf^1][row*32 + part*8]) = tmp[k];
          } else {
            int j = idx-128, d = j>>2, part = j&3;
            *(uint4*)(&sVal[buf^1][d*32 + part*8]) = tmp[k];
          }
        }
      }
    }
    __syncthreads();
#endif
  }

  // epilogue: normalize and write ctx (b, n, 176) fp32
  #pragma unroll
  for (int t=0;t<11;++t){
    #pragma unroll
    for (int i=0;i<8;++i){
      float o = acc[t][i] / l_run[i];
      ctx[((size_t)b*KN + rowBase + i + 8*hh)*176 + t*16 + r] = o;
    }
  }
}

// ====================== out paths + final projection ===========================
__global__ void __launch_bounds__(128) out_kernel(
    const float* __restrict__ hbuf, const float* __restrict__ ctx,
    const float* __restrict__ w3j, const float* __restrict__ w_out,
    const float* __restrict__ w_lin_out, float* __restrict__ outp)
{
  __shared__ float sH[176], sCx[176], sS[1352], sG[832], sO4[72], sO6[104], sW[512];
  const int tid = threadIdx.x;
  const int node = blockIdx.x;
  for (int e=tid; e<176; e+=128){
    sH[e]  = hbuf[(size_t)node*176 + e];
    sCx[e] = ctx [(size_t)node*176 + e];
  }
  for (int e=tid; e<72;  e+=128) sO4[e]=0.f;
  for (int e=tid; e<104; e+=128) sO6[e]=0.f;
  __syncthreads();
  for (int p=0; p<8; ++p){
    const int l1=c_pl1[6+p], l2=c_pl2[6+p], l3=c_pl3[6+p];
    const int ni=2*l1+1, nj=2*l2+1, nk=2*l3+1;
    const float* W = w3j + (6+p)*2197;
    for (int e=tid; e<512; e+=128) sW[e] = w_out[p*512 + e];
    for (int e=tid; e<KC*ni*nk; e+=128){          // S[v,i,k] = sum_j cx[v,j] W[i,j,k]
      int v=e/(ni*nk), rem=e%(ni*nk), i=rem/nk, k=rem%nk;
      const float* cx = (l2==4) ? (sCx + v*9) : (sCx + 72 + v*13);
      float s = 0.f;
      for (int j=0;j<nj;++j) s += cx[j]*W[(i*nj+j)*nk + k];
      sS[e] = s;
    }
    __syncthreads();
    for (int e=tid; e<KC*KC*nk; e+=128){          // G[u,v,k] = sum_i h[u,i] S[v,i,k]
      int u=e/(KC*nk), rem=e%(KC*nk), v=rem/nk, k=rem%nk;
      int hoff = u*22 + ((l1==4)?0:9);
      float g = 0.f;
      for (int i=0;i<ni;++i) g += sH[hoff+i]*sS[(v*ni+i)*nk + k];
      sG[e] = g;
    }
    __syncthreads();
    for (int e=tid; e<KC*nk; e+=128){             // o[w,k] += sum_{u,v} G wout
      int w=e/nk, k=e%nk;
      float o = 0.f;
      for (int u=0;u<KC;++u)
        for (int v=0;v<KC;++v)
          o += sG[(u*KC+v)*nk + k]*sW[(u*KC+v)*8 + w];
      if (l3==4) sO4[w*9+k] += o; else sO6[w*13+k] += o;
    }
    __syncthreads();
  }
  if (tid < 22){
    const float inv = 0.3535533905932738f; // 1/sqrt(C)
    float d = 0.f;
    if (tid < 9){ for (int u=0;u<KC;++u) d += sO4[u*9+tid]*w_lin_out[u]; }
    else { int j=tid-9; for (int u=0;u<KC;++u) d += sO6[u*13+j]*w_lin_out[8+u]; }
    outp[(size_t)node*22 + tid] = d*inv;
  }
}

// ====================== launch =================================================
extern "C" void kernel_launch(void* const* d_in, const int* in_sizes, int n_in,
                              void* d_out, int out_size, void* d_ws, size_t ws_size,
                              hipStream_t stream) {
  (void)in_sizes; (void)n_in; (void)out_size; (void)ws_size;
  const float* feat      = (const float*)d_in[0];
  const float* sh        = (const float*)d_in[1];
  const float* log_s     = (const float*)d_in[2];
  const float* pos_w     = (const float*)d_in[3];
  const float* pos_b     = (const float*)d_in[4];
  const float* w_lin_in  = (const float*)d_in[5];
  const float* w_val     = (const float*)d_in[6];
  const float* w_out     = (const float*)d_in[7];
  const float* w_lin_out = (const float*)d_in[8];

  char* ws = (char*)d_ws;
  float*     w3j   = (float*)    (ws + OFF_W3J);
  float*     pb    = (float*)    (ws + OFF_PB);
  _Float16*  q     = (_Float16*) (ws + OFF_Q);
  _Float16*  valsT = (_Float16*) (ws + OFF_VALST);
  float*     hbuf  = (float*)    (ws + OFF_H);
  float*     ctx   = (float*)    (ws + OFF_CTX);

  w3j_init_kernel<<<14, 256, 0, stream>>>(w3j);
  prep_kernel<<<KB*KN, 64, 0, stream>>>(feat, sh, log_s, pos_w, pos_b,
                                        w_lin_in, w_val, w3j, q, pb, valsT, hbuf);
  flash_kernel<<<dim3(KN/128, KB), 256, 0, stream>>>(q, pb, valsT, ctx);
  out_kernel<<<KB*KN, 128, 0, stream>>>(hbuf, ctx, w3j, w_out, w_lin_out, (float*)d_out);
}
